// ManualAttention_13417477833497
// MI455X (gfx1250) — matrix operand order repaired and checked by reading, not compiled, not run on hardware
//
#include <hip/hip_runtime.h>

typedef __attribute__((ext_vector_type(16))) __bf16 v16bf;
typedef __attribute__((ext_vector_type(8)))  __bf16 v8bf;
typedef __attribute__((ext_vector_type(4)))  __bf16 v4bf;
typedef __attribute__((ext_vector_type(8)))  float  v8f;
typedef __attribute__((ext_vector_type(8)))  unsigned int v8u;

constexpr int S_LEN  = 2048;
constexpr int D_DIM  = 128;
constexpr int BH_CNT = 64;      // B*H = 4*16
constexpr int KV_T   = 64;      // keys per iteration (4 score tiles of 16)
constexpr int N_IT   = S_LEN / KV_T;
constexpr int LDK    = 136;     // K LDS row stride (bf16): 128 + 8 pad  -> conflict-free b128 reads
constexpr int LDV    = 72;      // V^T LDS row stride (bf16): 64 + 8 pad -> conflict-free b128 reads

// Load a 16x32 bf16 A-fragment row-slice from an LDS row.
// A-layout (contiguous k order): lane holds k = {16h..16h+15}.
__device__ __forceinline__ v16bf lds_frag(const __bf16* rowbase, int h) {
  v8bf lo = *(const v8bf*)(rowbase + 16 * h);
  v8bf hi = *(const v8bf*)(rowbase + 16 * h + 8);
  return __builtin_shufflevector(lo, hi, 0,1,2,3,4,5,6,7,8,9,10,11,12,13,14,15);
}

__device__ __forceinline__ float f4e(const float4& f, int i) {
  return i == 0 ? f.x : i == 1 ? f.y : i == 2 ? f.z : f.w;
}

__device__ __forceinline__ unsigned pk2(float lo, float hi) {   // -> v_cvt_pk_bf16_f32
  unsigned short a = __builtin_bit_cast(unsigned short, (__bf16)lo);
  unsigned short b = __builtin_bit_cast(unsigned short, (__bf16)hi);
  return (unsigned)a | ((unsigned)b << 16);
}

// Online-softmax update of one 64-key score group; returns accumulator rescale.
__device__ __forceinline__ float softmax_update(v8f sT[4], float& m_run, float& l_run) {
  float mloc = sT[0][0];
  #pragma unroll
  for (int hh = 0; hh < 4; ++hh)
    #pragma unroll
    for (int r = 0; r < 8; ++r) mloc = fmaxf(mloc, sT[hh][r]);
  mloc = fmaxf(mloc, __shfl_xor(mloc, 16, 32));
  const float m_new = fmaxf(m_run, mloc);
  const float alpha = __builtin_amdgcn_exp2f(m_run - m_new);
  float lsum = 0.f;
  #pragma unroll
  for (int hh = 0; hh < 4; ++hh) {
    #pragma unroll
    for (int r = 0; r < 8; ++r) {
      sT[hh][r] = __builtin_amdgcn_exp2f(sT[hh][r] - m_new);
      lsum += sT[hh][r];
    }
  }
  lsum += __shfl_xor(lsum, 16, 32);
  l_run = l_run * alpha + lsum;
  m_run = m_new;
  return alpha;
}

// Build P^T B-fragments (j = 0..31 -> Pb0, 32..63 -> Pb1) from exp'd score tiles.
// Packed: cvt_pk pairs first, then one 32-bit xor16 shuffle + select per word.
__device__ __forceinline__ void build_pb(const v8f sT[4], bool h0, v16bf& Pb0, v16bf& Pb1) {
  v8u w0, w1;
  #pragma unroll
  for (int p = 0; p < 4; ++p) {
    unsigned own0 = pk2(sT[0][2*p], sT[0][2*p+1]);
    unsigned oth0 = pk2(sT[1][2*p], sT[1][2*p+1]);
    unsigned own1 = pk2(sT[2][2*p], sT[2][2*p+1]);
    unsigned oth1 = pk2(sT[3][2*p], sT[3][2*p+1]);
    unsigned s00 = (unsigned)__shfl_xor((int)oth0, 16, 32);
    unsigned s01 = (unsigned)__shfl_xor((int)own0, 16, 32);
    unsigned s10 = (unsigned)__shfl_xor((int)oth1, 16, 32);
    unsigned s11 = (unsigned)__shfl_xor((int)own1, 16, 32);
    w0[p]     = h0 ? own0 : s00;
    w0[4 + p] = h0 ? s01  : oth0;
    w1[p]     = h0 ? own1 : s10;
    w1[4 + p] = h0 ? s11  : oth1;
  }
  Pb0 = __builtin_bit_cast(v16bf, w0);
  Pb1 = __builtin_bit_cast(v16bf, w1);
}

__global__ __launch_bounds__(256)
void fa_wmma_kernel(const float* __restrict__ q, const float* __restrict__ k,
                    const float* __restrict__ v, float* __restrict__ out)
{
  __shared__ __bf16 Kb[KV_T * LDK];   // K tile, row-major [kv][d]      (17408 B)
  __shared__ __bf16 VT[D_DIM * LDV];  // V tile transposed [d][kv]      (18432 B)

  const int tid  = threadIdx.x;
  const int lane = tid & 31;
  const int wave = tid >> 5;
  const int h    = lane >> 4;
  const bool h0  = (h == 0);
  const int ln   = lane & 15;
  const int bh   = blockIdx.y;
  const int i0   = blockIdx.x * 256 + wave * 32;   // this wave's 32 Q rows (two 16-row groups)

  const float QSCALE = 0.08838834764831845f * 1.4426950408889634f; // 1/sqrt(D) * log2(e)

  // ---- preload Q^T B-fragments for both row groups (4 k-chunks each) ----
  v16bf Qta[4], Qtb[4];
  {
    const float* qa = q + ((size_t)bh * S_LEN + (i0 + ln)) * D_DIM;
    const float* qb = qa + 16 * D_DIM;
    #pragma unroll
    for (int c = 0; c < 4; ++c) {
      const float4* pa = (const float4*)(qa + 32 * c + 16 * h);
      const float4* pb = (const float4*)(qb + 32 * c + 16 * h);
      #pragma unroll
      for (int x = 0; x < 4; ++x) {
        float4 fa = pa[x], fb = pb[x];
        Qta[c][4*x+0] = (__bf16)(fa.x * QSCALE); Qta[c][4*x+1] = (__bf16)(fa.y * QSCALE);
        Qta[c][4*x+2] = (__bf16)(fa.z * QSCALE); Qta[c][4*x+3] = (__bf16)(fa.w * QSCALE);
        Qtb[c][4*x+0] = (__bf16)(fb.x * QSCALE); Qtb[c][4*x+1] = (__bf16)(fb.y * QSCALE);
        Qtb[c][4*x+2] = (__bf16)(fb.z * QSCALE); Qtb[c][4*x+3] = (__bf16)(fb.w * QSCALE);
      }
    }
  }

  v8f accA[8], accB[8];
  #pragma unroll
  for (int cc = 0; cc < 8; ++cc) {
    accA[cc] = (v8f){0.f,0.f,0.f,0.f,0.f,0.f,0.f,0.f};
    accB[cc] = (v8f){0.f,0.f,0.f,0.f,0.f,0.f,0.f,0.f};
  }
  float mA = -__builtin_inff(), lA = 0.f;
  float mB = -__builtin_inff(), lB = 0.f;

  const float* kbase = k + (size_t)bh * S_LEN * D_DIM;
  const float* vbase = v + (size_t)bh * S_LEN * D_DIM;
  const int krow = tid >> 2;          // K staging: kv row 0..63
  const int kseg = (tid & 3) * 32;    // K staging: 32-float d segment
  const int vr0  = (tid & 15) * 4;    // V staging: kv rows vr0..vr0+3
  const int vd0  = (tid >> 4) * 8;    // V staging: d window vd0..vd0+7

  for (int it = 0; it < N_IT; ++it) {
    const int kv0 = it * KV_T;
    __syncthreads();   // previous tile fully consumed

    // ---- stage K (row-major bf16) ----
    {
      const float4* gk = (const float4*)(kbase + (size_t)(kv0 + krow) * D_DIM + kseg);
      __bf16* kdst = &Kb[krow * LDK + kseg];
      #pragma unroll
      for (int x = 0; x < 8; ++x) {
        float4 f = gk[x];
        kdst[4*x+0] = (__bf16)f.x; kdst[4*x+1] = (__bf16)f.y;
        kdst[4*x+2] = (__bf16)f.z; kdst[4*x+3] = (__bf16)f.w;
      }
    }
    // ---- stage V transposed: 4 kv-adjacent bf16 per ds_store_b64, conflict-free ----
    {
      float4 va[4], vb4[4];
      #pragma unroll
      for (int rr = 0; rr < 4; ++rr) {
        const float* r = vbase + (size_t)(kv0 + vr0 + rr) * D_DIM + vd0;
        va[rr]  = ((const float4*)r)[0];
        vb4[rr] = ((const float4*)r)[1];
      }
      #pragma unroll
      for (int x = 0; x < 4; ++x) {
        v4bf p0 = { (__bf16)f4e(va[0],x),  (__bf16)f4e(va[1],x),
                    (__bf16)f4e(va[2],x),  (__bf16)f4e(va[3],x) };
        v4bf p1 = { (__bf16)f4e(vb4[0],x), (__bf16)f4e(vb4[1],x),
                    (__bf16)f4e(vb4[2],x), (__bf16)f4e(vb4[3],x) };
        *(v4bf*)(&VT[(vd0 + x)     * LDV + vr0]) = p0;
        *(v4bf*)(&VT[(vd0 + 4 + x) * LDV + vr0]) = p1;
      }
    }
    // ---- prefetch next tile into L2 while this tile is computed ----
    if (it + 1 < N_IT) {
      const int n0 = kv0 + KV_T;
      __builtin_prefetch(kbase + (size_t)(n0 + krow) * D_DIM + kseg, 0, 1);
      #pragma unroll
      for (int rr = 0; rr < 4; ++rr)
        __builtin_prefetch(vbase + (size_t)(n0 + vr0 + rr) * D_DIM + vd0, 0, 1);
    }
    __syncthreads();   // tile visible

    // ---- transposed scores for both row groups; K fragment reused 2x ----
    v8f sTa[4], sTb[4];
    #pragma unroll
    for (int hh = 0; hh < 4; ++hh) {
      sTa[hh] = (v8f){0.f,0.f,0.f,0.f,0.f,0.f,0.f,0.f};
      sTb[hh] = (v8f){0.f,0.f,0.f,0.f,0.f,0.f,0.f,0.f};
    }
    #pragma unroll
    for (int c = 0; c < 4; ++c) {
      #pragma unroll
      for (int hh = 0; hh < 4; ++hh) {
        v16bf a = lds_frag(&Kb[(16 * hh + ln) * LDK + 32 * c], h);
        sTa[hh] = __builtin_amdgcn_wmma_f32_16x16x32_bf16(false, a, false, Qta[c],
                                                          (short)0, sTa[hh], false, false);
        sTb[hh] = __builtin_amdgcn_wmma_f32_16x16x32_bf16(false, a, false, Qtb[c],
                                                          (short)0, sTb[hh], false, false);
      }
    }

    // ---- online softmax + P^T fragments, per row group ----
    const float alA = softmax_update(sTa, mA, lA);
    const float alB = softmax_update(sTb, mB, lB);
    v16bf Pb0a, Pb1a, Pb0b, Pb1b;
    build_pb(sTa, h0, Pb0a, Pb1a);
    build_pb(sTb, h0, Pb0b, Pb1b);

    // ---- rescale accumulators, then O^T += V^T x P^T ; V fragment reused 2x ----
    #pragma unroll
    for (int cc = 0; cc < 8; ++cc) {
      #pragma unroll
      for (int r = 0; r < 8; ++r) { accA[cc][r] *= alA; accB[cc][r] *= alB; }
    }
    #pragma unroll
    for (int cc = 0; cc < 8; ++cc) {
      const __bf16* vrow = &VT[(16 * cc + ln) * LDV];
      v16bf av0 = lds_frag(vrow, h);         // k = kv 0..31
      accA[cc] = __builtin_amdgcn_wmma_f32_16x16x32_bf16(false, av0, false, Pb0a,
                                                         (short)0, accA[cc], false, false);
      accB[cc] = __builtin_amdgcn_wmma_f32_16x16x32_bf16(false, av0, false, Pb0b,
                                                         (short)0, accB[cc], false, false);
      v16bf av1 = lds_frag(vrow + 32, h);    // k = kv 32..63
      accA[cc] = __builtin_amdgcn_wmma_f32_16x16x32_bf16(false, av1, false, Pb1a,
                                                         (short)0, accA[cc], false, false);
      accB[cc] = __builtin_amdgcn_wmma_f32_16x16x32_bf16(false, av1, false, Pb1b,
                                                         (short)0, accB[cc], false, false);
    }
  }

  // ---- epilogue: O(i,d) = O^T(d,i) / l, both row groups ----
  const float invA = 1.0f / lA;
  const float invB = 1.0f / lB;
  float* orowA = out + ((size_t)bh * S_LEN + (i0 + ln)) * D_DIM;
  float* orowB = orowA + 16 * D_DIM;
  #pragma unroll
  for (int cc = 0; cc < 8; ++cc) {
    float4 a0 = { accA[cc][0]*invA, accA[cc][1]*invA, accA[cc][2]*invA, accA[cc][3]*invA };
    float4 a1 = { accA[cc][4]*invA, accA[cc][5]*invA, accA[cc][6]*invA, accA[cc][7]*invA };
    float4 b0 = { accB[cc][0]*invB, accB[cc][1]*invB, accB[cc][2]*invB, accB[cc][3]*invB };
    float4 b1 = { accB[cc][4]*invB, accB[cc][5]*invB, accB[cc][6]*invB, accB[cc][7]*invB };
    float4* da = (float4*)(orowA + 16 * cc + 8 * h);
    float4* db = (float4*)(orowB + 16 * cc + 8 * h);
    da[0] = a0; da[1] = a1;
    db[0] = b0; db[1] = b1;
  }
}

extern "C" void kernel_launch(void* const* d_in, const int* in_sizes, int n_in,
                              void* d_out, int out_size, void* d_ws, size_t ws_size,
                              hipStream_t stream) {
  const float* q = (const float*)d_in[0];
  const float* k = (const float*)d_in[1];
  const float* v = (const float*)d_in[2];
  float* out = (float*)d_out;
  dim3 grid(S_LEN / 256, BH_CNT);   // 8 x 64 blocks
  dim3 block(256);                  // 8 waves; each wave owns 32 Q rows
  fa_wmma_kernel<<<grid, block, 0, stream>>>(q, k, v, out);
}
